// DenoisingHOGNN_21973052686529
// MI455X (gfx1250) — matrix operand order repaired and checked by reading, not compiled, not run on hardware
//
#include <hip/hip_runtime.h>
#include <hip/hip_bf16.h>

// MI455X / gfx1250, wave32, v_wmma_f32_16x16x32_f16.
// Workspace layout (needs >= 328MB):
//   [0,64MB)     X    (B,N,N,HID) f32 (residual stream, L2-resident)
//   [64,128MB)   XT   einsum out (b,d,i,j) f32 / ea (pre-loop) / Xs (post-loop)
//   [128,192MB)  T1   f16 mlp_ln stage-1 out / e1 (f16) / M1 (f16)
//   [192,256MB)  x1T  f16 WMMA-A-fragment-ordered / edge_h f32 (post-loop)
//   [256,320MB)  x2T  f16 WMMA-B-fragment-ordered
//   [320,321MB)  swizzled f16 weights (persistent)
//   [321MB..)    temb, node_h, t1, t2 (small)

typedef __attribute__((ext_vector_type(16))) _Float16 v16h;
typedef __attribute__((ext_vector_type(8)))  _Float16 v8h;
typedef __attribute__((ext_vector_type(8)))  float    v8f;

__device__ __forceinline__ float silu_f(float x) { return x / (1.0f + __expf(-x)); }

#define EPI_BIAS       0
#define EPI_SILU       1
#define EPI_LN_SILU    2
#define EPI_LN_SILU_TA 3   // write f16 in WMMA A-fragment order (b,d)[it][kc][lane][16]
#define EPI_MEANJ      4
#define EPI_LN_SILU_TB 5   // write f16 in WMMA B-fragment order (b,d)[jt][kc][lane][16]

// A fragment: lane holds row (lane&15); halves 0..7 = K[hi*8..+7], 8..15 = K[16+hi*8..+7]
template<typename TA>
__device__ __forceinline__ v16h load_a_frag(const TA* __restrict__ Arow, int k0, int hi)
{
  v16h a;
  if constexpr (sizeof(TA) == 2) {
    const v8h lo = *(const v8h*)((const _Float16*)Arow + k0 + hi * 8);
    const v8h hh = *(const v8h*)((const _Float16*)Arow + k0 + 16 + hi * 8);
    a = __builtin_shufflevector(lo, hh, 0, 1, 2, 3, 4, 5, 6, 7,
                                        8, 9, 10, 11, 12, 13, 14, 15);
  } else {
    const float* ap = (const float*)Arow + k0 + hi * 8;
    #pragma unroll
    for (int h = 0; h < 8; ++h) {
      a[h]     = (_Float16)ap[h];
      a[h + 8] = (_Float16)ap[16 + h];
    }
  }
  return a;
}

// Generic GEMM: out = epilogue(A(MxK) @ W(KxN) + bias). W pre-swizzled f16 fragments.
// 256 threads = 8 waves; block computes 16 rows x full N.
template<int EPI, typename TA, typename TO>
__global__ __launch_bounds__(256) void k_gemm(
    const TA* __restrict__ A, const _Float16* __restrict__ Wsw,
    const float* __restrict__ bias, const float* __restrict__ gamma,
    const float* __restrict__ beta, TO* __restrict__ out,
    int M, int N, int K)
{
  __shared__ float lds[16 * 256];
  const int tid  = threadIdx.x;
  const int wave = tid >> 5;
  const int lane = tid & 31;
  const int row  = lane & 15;
  const int hi   = lane >> 4;
  const int r0   = blockIdx.x * 16;
  const int KC   = K >> 5;

  const TA* Arow = A + (size_t)(r0 + row) * K;
  for (int nt = wave; nt < (N >> 4); nt += 8) {
    const v16h* Bf = (const v16h*)Wsw + (size_t)nt * KC * 32 + lane;
    v8f acc = {};
    for (int kc = 0; kc < KC; ++kc) {
      if (kc + 1 < KC)
        __builtin_prefetch((const void*)(Arow + (kc + 1) * 32), 0, 1); // global_prefetch_b8
      const v16h a = load_a_frag(Arow, kc * 32, hi);
      acc = __builtin_amdgcn_wmma_f32_16x16x32_f16(false, a, false, Bf[(size_t)kc * 32],
                                                   (short)0, acc, false, false);
    }
    #pragma unroll
    for (int r = 0; r < 8; ++r)
      lds[(r + hi * 8) * N + nt * 16 + row] = acc[r];
  }
  __syncthreads();

  if constexpr (EPI == EPI_BIAS || EPI == EPI_SILU) {
    for (int idx = tid; idx < 16 * N; idx += 256) {
      const int m = idx / N, n = idx % N;
      float v = lds[m * N + n] + bias[n];
      if constexpr (EPI == EPI_SILU) v = silu_f(v);
      out[(size_t)(r0 + m) * N + n] = (TO)v;
    }
  } else if constexpr (EPI == EPI_MEANJ) {
    // 16 consecutive rows share (b,i) since N2==128; accumulate mean over j.
    if (tid < N) {
      float s = 0.f;
      #pragma unroll
      for (int m = 0; m < 16; ++m) s += lds[m * N + tid] + bias[tid];
      atomicAdd((float*)&out[(size_t)(r0 >> 7) * N + tid], s * (1.0f / 128.0f));
    }
  } else {
    // LayerNorm over N==128 cols + SiLU. 16 threads per row (one 16-lane half).
    const int m   = tid >> 4;
    const int g16 = tid & 15;
    float vals[8], s1 = 0.f, s2 = 0.f;
    #pragma unroll
    for (int c = 0; c < 8; ++c) {
      const int n = g16 * 8 + c;
      const float v = lds[m * 128 + n] + bias[n];
      vals[c] = v; s1 += v; s2 += v * v;
    }
    #pragma unroll
    for (int off = 1; off < 16; off <<= 1) {   // stays inside 16-lane half
      s1 += __shfl_xor(s1, off, 32);
      s2 += __shfl_xor(s2, off, 32);
    }
    const float mean = s1 * (1.0f / 128.0f);
    const float var  = s2 * (1.0f / 128.0f) - mean * mean;
    const float rstd = rsqrtf(var + 1e-5f);
    #pragma unroll
    for (int c = 0; c < 8; ++c) {
      const int n = g16 * 8 + c;
      float v = (vals[c] - mean) * rstd * gamma[n] + beta[n];
      v = silu_f(v);
      if constexpr (EPI == EPI_LN_SILU) out[(size_t)(r0 + m) * 128 + n] = (TO)v;
      else                              lds[m * 128 + n] = v;
    }
    if constexpr (EPI == EPI_LN_SILU_TA) {
      __syncthreads();
      // rows = (b, i, kk0..kk0+15): emit A-fragments for batch p=(b,d).
      const int b = r0 >> 14, i = (r0 >> 7) & 127, kk0 = r0 & 127;
      const int it = i >> 4, kc = kk0 >> 5;
      const int h0 = (kk0 & 16) ? 8 : 0;
      const int d = tid >> 1, hhi = tid & 1;
      const int lane2 = hhi * 16 + (i & 15);
      v8h pk;
      #pragma unroll
      for (int c = 0; c < 8; ++c) pk[c] = (_Float16)lds[(hhi * 8 + c) * 128 + d];
      *(v8h*)((_Float16*)out +
              ((((size_t)(b * 128 + d) * 8 + it) * 4 + kc) * 32 + lane2) * 16 + h0) = pk;
    }
    if constexpr (EPI == EPI_LN_SILU_TB) {
      __syncthreads();
      // rows = (b, k, jj0..jj0+15): emit B-fragments for batch p=(b,d).
      const int b = r0 >> 14, k = (r0 >> 7) & 127, jj0 = r0 & 127;
      const int jt = jj0 >> 4, kc = k >> 5;
      const int r = k & 31, khi = (r >> 3) & 1, h = ((r >> 4) << 3) | (r & 7);
      #pragma unroll
      for (int ii = 0; ii < 8; ++ii) {
        const int idx = tid + ii * 256;
        const int d = idx >> 4, jm = idx & 15;
        ((_Float16*)out)[((((size_t)(b * 128 + d) * 8 + jt) * 4 + kc) * 32 +
                          (khi * 16 + jm)) * 16 + h] = (_Float16)lds[jm * 128 + d];
      }
    }
  }
}

// Pre-swizzle W(KxN f32) into f16 B-fragment order [nt][kc][lane][16].
__global__ void k_swz(const float* __restrict__ W, _Float16* __restrict__ out, int K, int N)
{
  const int idx = blockIdx.x * 256 + threadIdx.x;
  if (idx >= K * N) return;
  const int h = idx & 15, lane = (idx >> 4) & 31;
  const int t2 = idx >> 9, KC = K >> 5;
  const int kc = t2 % KC, nt = t2 / KC;
  const int k = kc * 32 + (h >> 3) * 16 + (lane >> 4) * 8 + (h & 7);
  const int n = nt * 16 + (lane & 15);
  out[idx] = (_Float16)W[(size_t)k * N + n];
}

// Batched einsum: C[p](128x128) = A[p] @ B[p], p = b*HID+d. Pure fragment streaming:
// one v16h A-load + one v16h B-load per v_wmma (both 32B/lane coalesced).
__global__ __launch_bounds__(256) void k_bmm(
    const _Float16* __restrict__ Afrag, const _Float16* __restrict__ Bfrag,
    float* __restrict__ C)
{
  const int tid = threadIdx.x, wave = tid >> 5, lane = tid & 31;
  const int row = lane & 15, hi = lane >> 4;
  const int tile = blockIdx.x * 8 + wave;
  const int p = tile >> 6, it = (tile >> 3) & 7, jt = tile & 7;
  const v16h* Af = (const v16h*)Afrag + ((size_t)p * 8 + it) * 128 + lane;
  const v16h* Bf = (const v16h*)Bfrag + ((size_t)p * 8 + jt) * 128 + lane;
  v8f acc = {};
  #pragma unroll
  for (int kc = 0; kc < 4; ++kc)
    acc = __builtin_amdgcn_wmma_f32_16x16x32_f16(false, Af[(size_t)kc * 32],
                                                 false, Bf[(size_t)kc * 32],
                                                 (short)0, acc, false, false);
  float* Cp = C + (size_t)p * 16384 + (size_t)it * 2048 + jt * 16;
  #pragma unroll
  for (int r = 0; r < 8; ++r)          // 16-lane 64B coalesced runs
    Cp[(size_t)(r + hi * 8) * 128 + row] = acc[r];
}

// X[b,i,j,d] += XT[b,d,i,j]  (LDS-tiled; both sides coalesce in 64B runs)
__global__ __launch_bounds__(256) void k_transpose_add(
    const float* __restrict__ XT, float* __restrict__ X)
{
  __shared__ float t[16][17];
  int tile = blockIdx.x;
  const int j0 = (tile & 7) * 16; tile >>= 3;
  const int d0 = (tile & 7) * 16; tile >>= 3;
  const int i  = tile & 127;
  const int b  = tile >> 7;
  const int tj = threadIdx.x & 15, td = threadIdx.x >> 4;
  t[td][tj] = XT[(((size_t)(b * 128 + d0 + td) * 128 + i) * 128) + j0 + tj];
  __syncthreads();
  const int d = threadIdx.x & 15, j = threadIdx.x >> 4;
  X[(((size_t)(b * 128 + i) * 128 + j0 + j) * 128) + d0 + d] += t[d][j];
}

// Xs[b,i,j,:] = X[b,i,j,:] + X[b,j,i,:]
__global__ void k_symm(const float* __restrict__ X, float* __restrict__ Xs)
{
  size_t q = (size_t)blockIdx.x * 256 + threadIdx.x;
  const int d4 = (int)(q & 31); q >>= 5;
  const int j  = (int)(q & 127); q >>= 7;
  const int i  = (int)(q & 127); q >>= 7;
  const int b  = (int)q;
  const float4 va = ((const float4*)(X + (((size_t)b * 128 + i) * 128 + j) * 128))[d4];
  const float4 vc = ((const float4*)(X + (((size_t)b * 128 + j) * 128 + i) * 128))[d4];
  ((float4*)(Xs + (((size_t)b * 128 + i) * 128 + j) * 128))[d4] =
      make_float4(va.x + vc.x, va.y + vc.y, va.z + vc.z, va.w + vc.w);
}

__global__ void k_zero(float* __restrict__ p, size_t n4)
{
  size_t q = (size_t)blockIdx.x * 256 + threadIdx.x;
  if (q < n4) ((float4*)p)[q] = make_float4(0.f, 0.f, 0.f, 0.f);
}

// X[g,si,di,:] += ea[e,:] * temb[g,:]
__global__ void k_scatter(const float* __restrict__ ea, const float* __restrict__ temb,
                          const int* __restrict__ ei, float* __restrict__ X)
{
  size_t q = (size_t)blockIdx.x * 256 + threadIdx.x;
  const int d4 = (int)(q & 31);
  const int e  = (int)(q >> 5);
  const int s  = ei[e], dv = ei[16384 + e];
  const int g = s >> 7, si = s & 127, di = dv & 127;
  const float4 v  = ((const float4*)(ea + (size_t)e * 128))[d4];
  const float4 tm = ((const float4*)(temb + (size_t)g * 128))[d4];
  float* dst = X + ((((size_t)g * 128 + si) * 128 + di) * 128) + (size_t)d4 * 4;
  atomicAdd(dst + 0, v.x * tm.x);
  atomicAdd(dst + 1, v.y * tm.y);
  atomicAdd(dst + 2, v.z * tm.z);
  atomicAdd(dst + 3, v.w * tm.w);
}

__global__ void k_gather(const float* __restrict__ Xs, const int* __restrict__ ei,
                         float* __restrict__ eh)
{
  size_t q = (size_t)blockIdx.x * 256 + threadIdx.x;
  const int d4 = (int)(q & 31);
  const int e  = (int)(q >> 5);
  const int s  = ei[e], dv = ei[16384 + e];
  const int g = s >> 7, si = s & 127, di = dv & 127;
  ((float4*)(eh + (size_t)e * 128))[d4] =
      ((const float4*)(Xs + (((size_t)g * 128 + si) * 128 + di) * 128))[d4];
}

// temb = silu(silu(t@W1+b1)@W2+b2), (8,128)
__global__ __launch_bounds__(256) void k_temb(
    const float* __restrict__ t, const float* __restrict__ W1, const float* __restrict__ b1,
    const float* __restrict__ W2, const float* __restrict__ b2, float* __restrict__ temb)
{
  __shared__ float h[1024];
  for (int idx = threadIdx.x; idx < 1024; idx += 256) {
    const int b = idx >> 7, d = idx & 127;
    h[idx] = silu_f(t[b] * W1[d] + b1[d]);
  }
  __syncthreads();
  for (int idx = threadIdx.x; idx < 1024; idx += 256) {
    const int b = idx >> 7, d = idx & 127;
    float s = b2[d];
    for (int k = 0; k < 128; ++k) s += h[b * 128 + k] * W2[k * 128 + d];
    temb[idx] = silu_f(s);
  }
}

__global__ void k_graph_mean(const float* __restrict__ node_out, float* __restrict__ gout)
{
  const int q = blockIdx.x * 256 + threadIdx.x;
  if (q >= 512) return;
  const int b = q >> 6, o = q & 63;
  float s = 0.f;
  for (int i = 0; i < 128; ++i) s += node_out[((size_t)b * 128 + i) * 64 + o];
  gout[q] = s * (1.0f / 128.0f);
}

extern "C" void kernel_launch(void* const* d_in, const int* in_sizes, int n_in,
                              void* d_out, int out_size, void* d_ws, size_t ws_size,
                              hipStream_t stream)
{
  const float* edge_attr = (const float*)d_in[1];
  const float* tval      = (const float*)d_in[2];
  const int*   ei        = (const int*)d_in[3];
  // d_in[4..7] = node_in_* : dead in the reference -> skipped.
  const float* einW1 = (const float*)d_in[8];
  const float* einb1 = (const float*)d_in[9];
  const float* einW2 = (const float*)d_in[10];
  const float* einb2 = (const float*)d_in[11];
  const float* tW1   = (const float*)d_in[12];
  const float* tb1   = (const float*)d_in[13];
  const float* tW2   = (const float*)d_in[14];
  const float* tb2   = (const float*)d_in[15];
  const float* gW    = (const float*)d_in[16];
  const float* gb    = (const float*)d_in[17];
  const float* gg    = (const float*)d_in[18];
  const float* gbe   = (const float*)d_in[19];
  const float* mW1   = (const float*)d_in[20];
  const float* mb1   = (const float*)d_in[21];
  const float* mW2   = (const float*)d_in[22];
  const float* mb2   = (const float*)d_in[23];
  const float* fnW1  = (const float*)d_in[24];
  const float* fnb1  = (const float*)d_in[25];
  const float* fnW2  = (const float*)d_in[26];
  const float* fnb2  = (const float*)d_in[27];
  const float* feW1  = (const float*)d_in[28];
  const float* feb1  = (const float*)d_in[29];
  const float* feW2  = (const float*)d_in[30];
  const float* feb2  = (const float*)d_in[31];

  char* ws = (char*)d_ws;
  const size_t MB = 1024ull * 1024ull;
  float*     X    = (float*)(ws + 0 * MB);
  float*     XT   = (float*)(ws + 64 * MB);
  _Float16*  T1h  = (_Float16*)(ws + 128 * MB);
  _Float16*  x1Th = (_Float16*)(ws + 192 * MB);
  _Float16*  x2Th = (_Float16*)(ws + 256 * MB);
  // overlapped (schedule-dead) aliases:
  float*     eaF    = XT;                               // pre-loop only
  _Float16*  e1h    = T1h;                              // pre-loop only
  float*     edge_h = (float*)(ws + 192 * MB);          // post-loop (x1T dead)
  // persistent/small:
  _Float16* swz   = (_Float16*)(ws + 320 * MB);         // 1MB swizzled weights
  float*    temb  = (float*)(ws + 321 * MB);
  float*    node_h= (float*)(ws + 322 * MB);
  _Float16* t1h   = (_Float16*)(ws + 323 * MB);
  _Float16* t2h   = (_Float16*)(ws + 324 * MB);         // 4MB -> needs ws >= 328MB

  _Float16* swEinW1 = swz;
  _Float16* swEinW2 = swz + 16384;
  _Float16* swGnn   = swz + 49152;       // 24 x 16384
  _Float16* swMW1   = swz + 442368;
  _Float16* swMW2   = swz + 458752;
  _Float16* swFnW1  = swz + 475136;
  _Float16* swFnW2  = swz + 491520;
  _Float16* swFeW1  = swz + 499712;
  _Float16* swFeW2  = swz + 516096;

  float* dout      = (float*)d_out;
  float* node_out  = dout;                 // 8*128*64
  float* edge_out  = dout + 65536;         // 16384*64
  float* graph_out = edge_out + 1048576;   // 8*64

  const int Mbig = 8 * 128 * 128;          // 131072

  // ---- weight pre-swizzle (f16 fragment order) ----
  k_swz<<<64, 256, 0, stream>>>(einW1, swEinW1, 64, 256);
  k_swz<<<128, 256, 0, stream>>>(einW2, swEinW2, 256, 128);
  for (int i = 0; i < 24; ++i)
    k_swz<<<64, 256, 0, stream>>>(gW + (size_t)i * 16384, swGnn + (size_t)i * 16384, 128, 128);
  k_swz<<<64, 256, 0, stream>>>(mW1, swMW1, 128, 128);
  k_swz<<<64, 256, 0, stream>>>(mW2, swMW2, 128, 128);
  k_swz<<<64, 256, 0, stream>>>(fnW1, swFnW1, 128, 128);
  k_swz<<<32, 256, 0, stream>>>(fnW2, swFnW2, 128, 64);
  k_swz<<<64, 256, 0, stream>>>(feW1, swFeW1, 128, 128);
  k_swz<<<32, 256, 0, stream>>>(feW2, swFeW2, 128, 64);

  // ---- time embedding + edge-in MLP ----
  k_temb<<<1, 256, 0, stream>>>(tval, tW1, tb1, tW2, tb2, temb);
  k_gemm<EPI_SILU, float, _Float16><<<1024, 256, 0, stream>>>(
      edge_attr, swEinW1, einb1, nullptr, nullptr, e1h, 16384, 256, 64);
  k_gemm<EPI_BIAS, _Float16, float><<<1024, 256, 0, stream>>>(
      e1h, swEinW2, einb2, nullptr, nullptr, eaF, 16384, 128, 256);

  // ---- X = scatter(ea * temb) ----
  k_zero<<<16384, 256, 0, stream>>>(X, 4194304);
  k_scatter<<<2048, 256, 0, stream>>>(eaF, temb, ei, X);

  // ---- L PPGN layers ----
  for (int l = 0; l < 6; ++l) {
    const size_t v = (size_t)l * 4 * 128;
    const _Float16 *W00 = swGnn + (size_t)(l * 4 + 0) * 16384,
                   *W01 = swGnn + (size_t)(l * 4 + 1) * 16384,
                   *W10 = swGnn + (size_t)(l * 4 + 2) * 16384,
                   *W11 = swGnn + (size_t)(l * 4 + 3) * 16384;
    const float *b00 = gb + v, *b01 = gb + v + 128, *b10 = gb + v + 256, *b11 = gb + v + 384;
    const float *g00 = gg + v, *g01 = gg + v + 128, *g10 = gg + v + 256, *g11 = gg + v + 384;
    const float *e00 = gbe + v, *e01 = gbe + v + 128, *e10 = gbe + v + 256, *e11 = gbe + v + 384;

    k_gemm<EPI_LN_SILU, float, _Float16><<<Mbig / 16, 256, 0, stream>>>(
        X, W00, b00, g00, e00, T1h, Mbig, 128, 128);
    k_gemm<EPI_LN_SILU_TA, _Float16, _Float16><<<Mbig / 16, 256, 0, stream>>>(
        T1h, W01, b01, g01, e01, x1Th, Mbig, 128, 128);
    k_gemm<EPI_LN_SILU, float, _Float16><<<Mbig / 16, 256, 0, stream>>>(
        X, W10, b10, g10, e10, T1h, Mbig, 128, 128);
    k_gemm<EPI_LN_SILU_TB, _Float16, _Float16><<<Mbig / 16, 256, 0, stream>>>(
        T1h, W11, b11, g11, e11, x2Th, Mbig, 128, 128);
    k_bmm<<<8192, 256, 0, stream>>>(x1Th, x2Th, XT);
    k_transpose_add<<<65536, 256, 0, stream>>>(XT, X);
  }

  // ---- X + X^T, heads, pools ----
  float* Xs = XT;
  k_symm<<<16384, 256, 0, stream>>>(X, Xs);
  k_gather<<<2048, 256, 0, stream>>>(Xs, ei, edge_h);

  k_gemm<EPI_SILU, float, _Float16><<<Mbig / 16, 256, 0, stream>>>(
      Xs, swMW1, mb1, nullptr, nullptr, T1h, Mbig, 128, 128);
  k_zero<<<128, 256, 0, stream>>>(node_h, 32768);
  k_gemm<EPI_MEANJ, _Float16, float><<<Mbig / 16, 256, 0, stream>>>(
      T1h, swMW2, mb2, nullptr, nullptr, node_h, Mbig, 128, 128);

  k_gemm<EPI_SILU, float, _Float16><<<64, 256, 0, stream>>>(
      node_h, swFnW1, fnb1, nullptr, nullptr, t1h, 1024, 128, 128);
  k_gemm<EPI_BIAS, _Float16, float><<<64, 256, 0, stream>>>(
      t1h, swFnW2, fnb2, nullptr, nullptr, node_out, 1024, 64, 128);

  k_gemm<EPI_SILU, float, _Float16><<<1024, 256, 0, stream>>>(
      edge_h, swFeW1, feb1, nullptr, nullptr, t2h, 16384, 128, 128);
  k_gemm<EPI_BIAS, _Float16, float><<<1024, 256, 0, stream>>>(
      t2h, swFeW2, feb2, nullptr, nullptr, edge_out, 16384, 64, 128);

  k_graph_mean<<<2, 256, 0, stream>>>(node_out, graph_out);
}
